// GRNN_kTop_32908039422444
// MI455X (gfx1250) — matrix operand order repaired and checked by reading, compiled, not run on hardware
//
#include <hip/hip_runtime.h>
#include <hip/hip_bf16.h>

// ---------------------------------------------------------------------------
// Types for CDNA5 WMMA (wave32, 16x16x32 f16 -> f32)
// ---------------------------------------------------------------------------
typedef _Float16 h8  __attribute__((ext_vector_type(8)));
typedef _Float16 v16h __attribute__((ext_vector_type(16)));
typedef float    v8f  __attribute__((ext_vector_type(8)));

#define LDS_STRIDE 40   // halfs per row (80B, 16B-aligned, padded vs 32 to break bank conflicts)

// ---------------------------------------------------------------------------
// gfx1250 async global->LDS copy (ASYNCcnt-tracked, no VGPR round trip).
// ISA 15.18.3 opcode 98: copies 16 bytes per lane from global memory into LDS.
// ---------------------------------------------------------------------------
__device__ __forceinline__ void async_b128(unsigned lds_off, const void* g) {
  asm volatile("global_load_async_to_lds_b128 %0, %1, off"
               :: "v"(lds_off), "v"(g) : "memory");
}
__device__ __forceinline__ void async_wait0() {
  asm volatile("s_wait_asynccnt 0x0" ::: "memory");
}

__device__ __forceinline__ v16h load_frag_a(const _Float16* lds, int row, int laneHi) {
  // 16-bit A 16x32 layout: lanes0-15 hold K 0..7 / 16..23, lanes16-31 hold K 8..15 / 24..31
  const _Float16* r = lds + row * LDS_STRIDE;
  h8 lo = *(const h8*)(r + (laneHi ? 8 : 0));
  h8 hi = *(const h8*)(r + 16 + (laneHi ? 8 : 0));
  v16h f;
#pragma unroll
  for (int x = 0; x < 8; x++) { f[x] = lo[x]; f[8 + x] = hi[x]; }
  return f;
}

__device__ __forceinline__ v16h load_frag_b(const _Float16* lds, int col, int laneHi) {
  // B stored N-major in LDS: lds[col][k]; lanes0-15 take K 0..7 / 16..23, lanes16-31 take K 8..15 / 24..31
  const _Float16* r = lds + col * LDS_STRIDE + (laneHi ? 8 : 0);
  h8 lo = *(const h8*)(r);
  h8 hi = *(const h8*)(r + 16);
  v16h f;
#pragma unroll
  for (int x = 0; x < 8; x++) { f[x] = lo[x]; f[8 + x] = hi[x]; }
  return f;
}

// ---------------------------------------------------------------------------
// Generic NT GEMM:  C(MxN,f32) = A(MxK,f16) @ W(NxK,f16)^T  [+bias[n]] [act]
// K must be a multiple of 32. Block = 256 thr (8 waves), tile 128x128.
// A/W tiles staged via async global->LDS (rows OOB pre-zeroed once).
// ---------------------------------------------------------------------------
__global__ __launch_bounds__(256)
void gemm_nt(const _Float16* __restrict__ A, int lda,
             const _Float16* __restrict__ W, int ldw,
             float* __restrict__ C, int ldc,
             int M, int N, int K,
             const float* __restrict__ bias, int act) {
  __shared__ _Float16 ldsA[128 * LDS_STRIDE];
  __shared__ _Float16 ldsB[128 * LDS_STRIDE];
  const int tid = threadIdx.x;
  const int lane = tid & 31, wave = tid >> 5;
  const int laneLo = lane & 15, laneHi = lane >> 4;
  const int wm = wave >> 1, wn = wave & 1;
  const int m0 = blockIdx.x * 128, n0 = blockIdx.y * 128;
  const int lr = tid >> 1;
  const int seg = (tid & 1) * 16;
  const unsigned baseA = (unsigned)(size_t)(void*)ldsA;
  const unsigned baseB = (unsigned)(size_t)(void*)ldsB;
  const bool aOk = (m0 + lr) < M;
  const bool bOk = (n0 + lr) < N;

  v8f acc[2][4];
#pragma unroll
  for (int i = 0; i < 2; i++)
#pragma unroll
    for (int j = 0; j < 4; j++)
#pragma unroll
      for (int v = 0; v < 8; v++) acc[i][j][v] = 0.f;

  h8 z8;
#pragma unroll
  for (int x = 0; x < 8; x++) z8[x] = (_Float16)0.f;

  // Pre-zero rows outside M/N once (K-invariant predicate), so async loads can
  // simply be skipped on those rows.
  if (!aOk) {
    *(h8*)(&ldsA[lr * LDS_STRIDE + seg]) = z8;
    *(h8*)(&ldsA[lr * LDS_STRIDE + seg + 8]) = z8;
  }
  if (!bOk) {
    *(h8*)(&ldsB[lr * LDS_STRIDE + seg]) = z8;
    *(h8*)(&ldsB[lr * LDS_STRIDE + seg + 8]) = z8;
  }
  __syncthreads();

  const unsigned la = baseA + (unsigned)((lr * LDS_STRIDE + seg) * 2);
  const unsigned lb = baseB + (unsigned)((lr * LDS_STRIDE + seg) * 2);

  for (int kb = 0; kb < K; kb += 32) {
    if (aOk) {
      const _Float16* src = A + (size_t)(m0 + lr) * lda + kb + seg;
      async_b128(la, src);
      async_b128(la + 16, src + 8);
    }
    if (bOk) {
      const _Float16* src = W + (size_t)(n0 + lr) * ldw + kb + seg;
      async_b128(lb, src);
      async_b128(lb + 16, src + 8);
    }
    async_wait0();
    __syncthreads();
    v16h af[2], bf[4];
#pragma unroll
    for (int i = 0; i < 2; i++) af[i] = load_frag_a(ldsA, wm * 32 + i * 16 + laneLo, laneHi);
#pragma unroll
    for (int j = 0; j < 4; j++) bf[j] = load_frag_b(ldsB, wn * 64 + j * 16 + laneLo, laneHi);
#pragma unroll
    for (int i = 0; i < 2; i++)
#pragma unroll
      for (int j = 0; j < 4; j++)
        acc[i][j] = __builtin_amdgcn_wmma_f32_16x16x32_f16(
            false, af[i], false, bf[j], (short)0, acc[i][j], false, false);
    __syncthreads();
  }

#pragma unroll
  for (int i = 0; i < 2; i++)
#pragma unroll
    for (int j = 0; j < 4; j++)
#pragma unroll
      for (int v = 0; v < 8; v++) {
        int row = m0 + wm * 32 + i * 16 + laneHi * 8 + v;
        int col = n0 + wn * 64 + j * 16 + laneLo;
        if (row < M && col < N) {
          float y = acc[i][j][v];
          if (bias) y += bias[col];
          if (act == 1) y = y > 0.f ? y : 0.1f * y;
          C[(size_t)row * ldc + col] = y;
        }
      }
}

// ---------------------------------------------------------------------------
// Conv1d(k=5,pad=2) over channels as 5 shifted WMMA GEMMs, all T timesteps.
// X,Y: (T,1024,128) f16. Wk: (5,1024,1024) f16. Y = leaky(conv(X)+b).
// Block tile 256(M)x128(N); 8 waves in 4x2, wave tile 64x64 -> 16 WMMA/K-step.
// Weight tiles streamed with async global->LDS. Grid: (4 Mtiles, T).
// ---------------------------------------------------------------------------
__global__ __launch_bounds__(256)
void conv_gemm(const _Float16* __restrict__ X,
               const _Float16* __restrict__ Wk,
               const float* __restrict__ bias,
               _Float16* __restrict__ Y) {
  __shared__ _Float16 ldsA[256 * LDS_STRIDE];
  __shared__ _Float16 ldsB[128 * LDS_STRIDE];
  const int tid = threadIdx.x;
  const int lane = tid & 31, wave = tid >> 5;
  const int laneLo = lane & 15, laneHi = lane >> 4;
  const int wm = wave >> 1;   // 0..3 : 64-row slice
  const int wn = wave & 1;    // 0..1 : 64-col slice
  const int t = blockIdx.y;
  const int m0 = blockIdx.x * 256;
  const int lr = tid >> 1;
  const int seg = (tid & 1) * 16;
  const _Float16* Xt = X + (size_t)t * 131072;
  const unsigned baseA = (unsigned)(size_t)(void*)ldsA;
  const unsigned la = baseA + (unsigned)(tid * (LDS_STRIDE * 2));  // one 32-half row per thread

  v8f acc[4][4];
#pragma unroll
  for (int i = 0; i < 4; i++)
#pragma unroll
    for (int j = 0; j < 4; j++)
#pragma unroll
      for (int v = 0; v < 8; v++) acc[i][j][v] = 0.f;

  h8 z8;
#pragma unroll
  for (int x = 0; x < 8; x++) z8[x] = (_Float16)0.f;

  for (int tap = 0; tap < 5; tap++) {
    const int shift = tap - 2;
    const _Float16* Wt = Wk + (size_t)tap * 1048576;
    for (int kb = 0; kb < 1024; kb += 32) {
      {  // A tile: 256 weight rows x 32 K; thread tid streams its whole row async
        const _Float16* src = Wt + (size_t)(m0 + tid) * 1024 + kb;
        async_b128(la, src);
        async_b128(la + 16, src + 8);
        async_b128(la + 32, src + 16);
        async_b128(la + 48, src + 24);
      }
      {  // B tile: ldsB[n][k] = X[kb+k][n+shift], zero-pad at sequence edges
        int n = lr;
        int l = n + shift;
        if (l >= 0 && l < 128) {
#pragma unroll
          for (int kk = 0; kk < 16; kk++)
            ldsB[n * LDS_STRIDE + seg + kk] = Xt[(size_t)(kb + seg + kk) * 128 + l];
        } else {
          *(h8*)(&ldsB[n * LDS_STRIDE + seg]) = z8;
          *(h8*)(&ldsB[n * LDS_STRIDE + seg + 8]) = z8;
        }
      }
      async_wait0();
      __syncthreads();
      v16h af[4], bf[4];
#pragma unroll
      for (int i = 0; i < 4; i++) af[i] = load_frag_a(ldsA, wm * 64 + i * 16 + laneLo, laneHi);
#pragma unroll
      for (int j = 0; j < 4; j++) bf[j] = load_frag_b(ldsB, wn * 64 + j * 16 + laneLo, laneHi);
#pragma unroll
      for (int i = 0; i < 4; i++)
#pragma unroll
        for (int j = 0; j < 4; j++)
          acc[i][j] = __builtin_amdgcn_wmma_f32_16x16x32_f16(
              false, af[i], false, bf[j], (short)0, acc[i][j], false, false);
      __syncthreads();
    }
  }

#pragma unroll
  for (int i = 0; i < 4; i++)
#pragma unroll
    for (int j = 0; j < 4; j++)
#pragma unroll
      for (int v = 0; v < 8; v++) {
        int o = m0 + wm * 64 + i * 16 + laneHi * 8 + v;
        int l = wn * 64 + j * 16 + laneLo;
        float y = acc[i][j][v] + bias[o];
        y = y > 0.f ? y : 0.1f * y;
        Y[(size_t)t * 131072 + (size_t)o * 128 + l] = (_Float16)y;
      }
}

// ---------------------------------------------------------------------------
// Small helper kernels
// ---------------------------------------------------------------------------
__global__ void cvt_f16(const float* __restrict__ s, _Float16* __restrict__ d, int n) {
  int i = blockIdx.x * blockDim.x + threadIdx.x;
  if (i < n) d[i] = (_Float16)s[i];
}

// (O=1024, I=1024, 5) f32  ->  (5, O, I) f16
__global__ void conv_w_repack(const float* __restrict__ w, _Float16* __restrict__ out) {
  int idx = blockIdx.x * blockDim.x + threadIdx.x;
  if (idx >= 5 * 1024 * 1024) return;
  int k = idx >> 20;
  int rem = idx & 0xFFFFF;
  int o = rem >> 10;
  int i = rem & 1023;
  out[idx] = (_Float16)w[(size_t)(o * 1024 + i) * 5 + k];
}

// (R,C) f32 -> (Rp,Cp) f16 zero-padded
__global__ void pad_repack_f16(const float* __restrict__ src, _Float16* __restrict__ dst,
                               int R, int C, int Rp, int Cp) {
  int i = blockIdx.x * blockDim.x + threadIdx.x;
  if (i >= Rp * Cp) return;
  int r = i / Cp, c = i % Cp;
  dst[i] = (r < R && c < C) ? (_Float16)src[(size_t)r * C + c] : (_Float16)0.f;
}

// (R,C) f32 -> (C,R) f16 transpose
__global__ void transpose_f16(const float* __restrict__ src, _Float16* __restrict__ dst,
                              int R, int C) {
  int i = blockIdx.x * blockDim.x + threadIdx.x;
  if (i >= R * C) return;
  int r = i / C, c = i % C;
  dst[(size_t)c * R + r] = (_Float16)src[i];
}

// GRU gates; gi/gh are (1024 x 304) f32, h is (1024 x 100)
__global__ void gru_gate(const float* __restrict__ gi, const float* __restrict__ gh,
                         const float* __restrict__ hprev, float* __restrict__ hnew,
                         _Float16* __restrict__ h16, _Float16* __restrict__ tfeat, int t) {
  int i = blockIdx.x * blockDim.x + threadIdx.x;
  if (i >= 1024 * 100) return;
  int node = i / 100, j = i % 100;
  const float* gin = gi + (size_t)node * 304;
  const float* ghn = gh + (size_t)node * 304;
  float r = 1.f / (1.f + __expf(-(gin[j] + ghn[j])));
  float z = 1.f / (1.f + __expf(-(gin[100 + j] + ghn[100 + j])));
  float nn = tanhf(gin[200 + j] + r * ghn[200 + j]);
  float hp = hprev[(size_t)node * 100 + j];
  float h = (1.f - z) * nn + z * hp;
  hnew[(size_t)node * 100 + j] = h;
  h16[(size_t)node * 128 + j] = (_Float16)h;
  float lh = h > 0.f ? h : 0.1f * h;                       // _leaky(t)
  tfeat[(size_t)node * 3200 + t * 100 + j] = (_Float16)lh;
}

__global__ void rowsq(const _Float16* __restrict__ x, float* __restrict__ sq, int n, int k) {
  int i = blockIdx.x * blockDim.x + threadIdx.x;
  if (i >= n) return;
  float s = 0.f;
  for (int c = 0; c < k; c++) { float v = (float)x[(size_t)i * k + c]; s += v * v; }
  sq[i] = s;
}

// one wave per row; d2 = sq_i + sq_j - 2*dot; select 20 smallest indices
__global__ void topk_from_dot(const float* __restrict__ dot, const float* __restrict__ sq,
                              int* __restrict__ idx, int n) {
  int row = blockIdx.x;
  int lane = threadIdx.x;
  float v[32];
  float sr = sq[row];
  for (int c = 0; c < 32; c++) {
    int j = c * 32 + lane;
    v[c] = (j < n) ? (sr + sq[j] - 2.f * dot[(size_t)row * n + j]) : 3.4e38f;
  }
  for (int it = 0; it < 20; it++) {
    float best = 3.4e38f; int bc = 0;
    for (int c = 0; c < 32; c++) if (v[c] < best) { best = v[c]; bc = c; }
    int bj = bc * 32 + lane;
    for (int off = 16; off > 0; off >>= 1) {
      float ov = __shfl_xor(best, off, 32);
      int oj = __shfl_xor(bj, off, 32);
      if (ov < best || (ov == best && oj < bj)) { best = ov; bj = oj; }
    }
    if (lane == 0) idx[row * 20 + it] = bj;
    if ((bj & 31) == lane) v[bj >> 5] = 3.4e38f;
  }
}

// one wave per row; distances computed directly from (n x 16) features
__global__ void topk_direct(const float* __restrict__ x, int* __restrict__ idx, int n) {
  int row = blockIdx.x;
  int lane = threadIdx.x;
  float xr[16];
#pragma unroll
  for (int c = 0; c < 16; c++) xr[c] = x[(size_t)row * 16 + c];
  float v[32];
  for (int c = 0; c < 32; c++) {
    int j = c * 32 + lane;
    float d = 3.4e38f;
    if (j < n) {
      d = 0.f;
#pragma unroll
      for (int q = 0; q < 16; q++) { float t = xr[q] - x[(size_t)j * 16 + q]; d += t * t; }
    }
    v[c] = d;
  }
  for (int it = 0; it < 20; it++) {
    float best = 3.4e38f; int bc = 0;
    for (int c = 0; c < 32; c++) if (v[c] < best) { best = v[c]; bc = c; }
    int bj = bc * 32 + lane;
    for (int off = 16; off > 0; off >>= 1) {
      float ov = __shfl_xor(best, off, 32);
      int oj = __shfl_xor(bj, off, 32);
      if (ov < best || (ov == best && oj < bj)) { best = ov; bj = oj; }
    }
    if (lane == 0) idx[row * 20 + it] = bj;
    if ((bj & 31) == lane) v[bj >> 5] = 3.4e38f;
  }
}

// h[i,j] = relu(P[i]+Q[idx]-Q[i]+b1) @ w2 + b2; max over 20 neighbors
__global__ void edge_mlp_max(const float* __restrict__ P, const float* __restrict__ Q,
                             const int* __restrict__ idx, const float* __restrict__ b1,
                             const float* __restrict__ w2, const float* __restrict__ b2,
                             float* __restrict__ out, int n, int finalRelu) {
  int i = blockIdx.x * blockDim.x + threadIdx.x;
  if (i >= n) return;
  float p[16], qi[16], bb[16], best[16];
#pragma unroll
  for (int c = 0; c < 16; c++) {
    p[c] = P[(size_t)i * 16 + c];
    qi[c] = Q[(size_t)i * 16 + c];
    bb[c] = b1[c];
    best[c] = -3.4e38f;
  }
  for (int e = 0; e < 20; e++) {
    int j = idx[i * 20 + e];
    float v[16];
#pragma unroll
    for (int c = 0; c < 16; c++) {
      float t = p[c] + Q[(size_t)j * 16 + c] - qi[c] + bb[c];
      v[c] = t > 0.f ? t : 0.f;
    }
#pragma unroll
    for (int c2 = 0; c2 < 16; c2++) {
      float o = b2[c2];
#pragma unroll
      for (int c = 0; c < 16; c++) o += v[c] * w2[c * 16 + c2];
      best[c2] = fmaxf(best[c2], o);
    }
  }
#pragma unroll
  for (int c = 0; c < 16; c++)
    out[(size_t)i * 16 + c] = finalRelu ? fmaxf(best[c], 0.f) : best[c];
}

// P2/Q2 projections for the 16-dim DGCN (w is (32,16) row-major)
__global__ void proj16(const float* __restrict__ x, const float* __restrict__ w,
                       float* __restrict__ P, float* __restrict__ Q, int n) {
  int i = blockIdx.x * blockDim.x + threadIdx.x;
  if (i >= n) return;
  float xi[16];
#pragma unroll
  for (int c = 0; c < 16; c++) xi[c] = x[(size_t)i * 16 + c];
#pragma unroll
  for (int c2 = 0; c2 < 16; c2++) {
    float sp = 0.f, sq = 0.f;
#pragma unroll
    for (int k = 0; k < 16; k++) {
      sp += xi[k] * w[k * 16 + c2];
      sq += xi[k] * w[(16 + k) * 16 + c2];
    }
    P[(size_t)i * 16 + c2] = sp;
    Q[(size_t)i * 16 + c2] = sq;
  }
}

__global__ void pool_score(const float* __restrict__ x, const float* __restrict__ w,
                           float* __restrict__ s, int n) {
  int i = blockIdx.x * blockDim.x + threadIdx.x;
  if (i >= n) return;
  float nw = 0.f, wl[16];
#pragma unroll
  for (int c = 0; c < 16; c++) { wl[c] = w[c]; nw += wl[c] * wl[c]; }
  nw = sqrtf(nw);
  float d = 0.f;
#pragma unroll
  for (int c = 0; c < 16; c++) d += x[(size_t)i * 16 + c] * wl[c];
  s[i] = tanhf(d / nw);
}

// rank-compaction keep (order-invariant downstream: only selected set matters)
__global__ void pool_compact(const float* __restrict__ x, const float* __restrict__ s,
                             float* __restrict__ xo, int n, int keep) {
  int i = blockIdx.x * blockDim.x + threadIdx.x;
  if (i >= n) return;
  float si = s[i];
  int rank = 0;
  for (int m = 0; m < n; m++) {
    float sm = s[m];
    if (sm > si || (sm == si && m < i)) rank++;
  }
  if (rank < keep) {
#pragma unroll
    for (int c = 0; c < 16; c++) xo[(size_t)rank * 16 + c] = x[(size_t)i * 16 + c] * si;
  }
}

// softmax-attention global pool -> 16 floats
__global__ void gap_k(const float* __restrict__ x, int n, const float* __restrict__ gw,
                      const float* __restrict__ gb, float* __restrict__ out) {
  __shared__ float red[256];
  __shared__ float res[17];
  int tid = threadIdx.x;
  float wl[16];
#pragma unroll
  for (int c = 0; c < 16; c++) wl[c] = gw[c];
  float gbv = gb[0];
  float lmax = -3.4e38f;
  for (int i = tid; i < n; i += 256) {
    float s = gbv;
#pragma unroll
    for (int c = 0; c < 16; c++) s += x[(size_t)i * 16 + c] * wl[c];
    lmax = fmaxf(lmax, s);
  }
  red[tid] = lmax; __syncthreads();
  for (int o = 128; o > 0; o >>= 1) { if (tid < o) red[tid] = fmaxf(red[tid], red[tid + o]); __syncthreads(); }
  float gmax = red[0]; __syncthreads();
  float lsum = 0.f, lacc[16];
#pragma unroll
  for (int c = 0; c < 16; c++) lacc[c] = 0.f;
  for (int i = tid; i < n; i += 256) {
    float s = gbv;
#pragma unroll
    for (int c = 0; c < 16; c++) s += x[(size_t)i * 16 + c] * wl[c];
    float e = __expf(s - gmax);
    lsum += e;
#pragma unroll
    for (int c = 0; c < 16; c++) lacc[c] += e * x[(size_t)i * 16 + c];
  }
  for (int v = 0; v < 17; v++) {
    red[tid] = (v == 0) ? lsum : lacc[v - 1];
    __syncthreads();
    for (int o = 128; o > 0; o >>= 1) { if (tid < o) red[tid] += red[tid + o]; __syncthreads(); }
    if (tid == 0) res[v] = red[0];
    __syncthreads();
  }
  if (tid == 0) {
#pragma unroll
    for (int c = 0; c < 16; c++) out[c] = res[1 + c] / res[0];
  }
}

__global__ void final_k(const float* __restrict__ g1v, const float* __restrict__ g2v,
                        const float* __restrict__ w, const float* __restrict__ b,
                        float* __restrict__ out) {
  int c = threadIdx.x;
  if (c < 2) {
    float s = b[c];
#pragma unroll
    for (int k = 0; k < 16; k++) s += (g1v[k] + g2v[k]) * w[k * 2 + c];
    out[c] = s;
  }
}

// ---------------------------------------------------------------------------
// Host orchestration
// ---------------------------------------------------------------------------
extern "C" void kernel_launch(void* const* d_in, const int* in_sizes, int n_in,
                              void* d_out, int out_size, void* d_ws, size_t ws_size,
                              hipStream_t stream) {
  const float* ts      = (const float*)d_in[0];
  const float* conv_w[3] = {(const float*)d_in[1], (const float*)d_in[3], (const float*)d_in[5]};
  const float* conv_b[3] = {(const float*)d_in[2], (const float*)d_in[4], (const float*)d_in[6]};
  const float* gru_wih = (const float*)d_in[7];
  const float* gru_whh = (const float*)d_in[8];
  const float* gru_bih = (const float*)d_in[9];
  const float* gru_bhh = (const float*)d_in[10];
  const float* mlp1_w1 = (const float*)d_in[11];
  const float* mlp1_b1 = (const float*)d_in[12];
  const float* mlp1_w2 = (const float*)d_in[13];
  const float* mlp1_b2 = (const float*)d_in[14];
  const float* pool1_w = (const float*)d_in[15];
  const float* gate1_w = (const float*)d_in[16];
  const float* gate1_b = (const float*)d_in[17];
  const float* mlp2_w1 = (const float*)d_in[18];
  const float* mlp2_b1 = (const float*)d_in[19];
  const float* mlp2_w2 = (const float*)d_in[20];
  const float* mlp2_b2 = (const float*)d_in[21];
  const float* pool2_w = (const float*)d_in[22];
  const float* gate2_w = (const float*)d_in[23];
  const float* gate2_b = (const float*)d_in[24];
  const float* lin2_w  = (const float*)d_in[25];
  const float* lin2_b  = (const float*)d_in[26];

  char* base = (char*)d_ws;
  size_t off = 0;
  auto al = [&](size_t bytes) -> void* {
    void* r = base + off;
    off = (off + bytes + 255) & ~(size_t)255;
    return r;
  };

  _Float16* bufA  = (_Float16*)al(32ull * 1024 * 128 * 2);   // X0 f16 / conv3 out
  _Float16* bufB  = (_Float16*)al(32ull * 1024 * 128 * 2);   // conv1 out
  _Float16* bufC  = (_Float16*)al(32ull * 1024 * 128 * 2);   // conv2 out
  _Float16* wconv = (_Float16*)al(5ull * 1024 * 1024 * 2);   // repacked conv weights (reused per layer)
  _Float16* wih16 = (_Float16*)al(304 * 128 * 2);
  _Float16* whh16 = (_Float16*)al(304 * 128 * 2);
  float*    gi    = (float*)al(1024 * 304 * 4);
  float*    gh    = (float*)al(1024 * 304 * 4);
  float*    hA    = (float*)al(1024 * 100 * 4);
  float*    hB    = (float*)al(1024 * 100 * 4);
  _Float16* h16   = (_Float16*)al(1024 * 128 * 2);
  _Float16* tfeat = (_Float16*)al(1024ull * 3200 * 2);
  float*    xxT   = (float*)al(1024ull * 1024 * 4);
  float*    sq    = (float*)al(1024 * 4);
  int*      idx1  = (int*)al(1024 * 20 * 4);
  _Float16* w1t   = (_Float16*)al(16 * 6400 * 2);
  float*    P1    = (float*)al(1024 * 16 * 4);
  float*    Q1    = (float*)al(1024 * 16 * 4);
  float*    xdg1  = (float*)al(1024 * 16 * 4);
  float*    sc1   = (float*)al(1024 * 4);
  float*    xp1   = (float*)al(820 * 16 * 4);
  int*      idx2  = (int*)al(820 * 20 * 4);
  float*    P2    = (float*)al(820 * 16 * 4);
  float*    Q2    = (float*)al(820 * 16 * 4);
  float*    xdg2  = (float*)al(820 * 16 * 4);
  float*    sc2   = (float*)al(820 * 4);
  float*    xp2   = (float*)al(656 * 16 * 4);
  float*    g1v   = (float*)al(16 * 4);
  float*    g2v   = (float*)al(16 * 4);
  (void)ws_size; (void)n_in; (void)in_sizes; (void)out_size;

  // --- timeseries -> f16 ---
  cvt_f16<<<(32 * 1024 * 128 + 255) / 256, 256, 0, stream>>>(ts, bufA, 32 * 1024 * 128);

  // --- 3 conv layers (WMMA), all 32 timesteps batched per layer ---
  _Float16* cin[3]  = {bufA, bufB, bufC};
  _Float16* cout[3] = {bufB, bufC, bufA};
  for (int L = 0; L < 3; L++) {
    conv_w_repack<<<(5 * 1024 * 1024 + 255) / 256, 256, 0, stream>>>(conv_w[L], wconv);
    conv_gemm<<<dim3(4, 32), 256, 0, stream>>>(cin[L], wconv, conv_b[L], cout[L]);
  }
  _Float16* Y3 = bufA;  // conv3 output (T,1024,128) f16

  // --- GRU (32 sequential steps; WMMA projections + scalar gates) ---
  pad_repack_f16<<<(304 * 128 + 255) / 256, 256, 0, stream>>>(gru_wih, wih16, 300, 128, 304, 128);
  pad_repack_f16<<<(304 * 128 + 255) / 256, 256, 0, stream>>>(gru_whh, whh16, 300, 100, 304, 128);
  hipMemsetAsync(hA, 0, 1024 * 100 * 4, stream);
  hipMemsetAsync(h16, 0, 1024 * 128 * 2, stream);
  float* hp = hA;
  float* hn = hB;
  for (int t = 0; t < 32; t++) {
    gemm_nt<<<dim3(8, 3), 256, 0, stream>>>(Y3 + (size_t)t * 131072, 128, wih16, 128,
                                            gi, 304, 1024, 300, 128, gru_bih, 0);
    gemm_nt<<<dim3(8, 3), 256, 0, stream>>>(h16, 128, whh16, 128,
                                            gh, 304, 1024, 300, 128, gru_bhh, 0);
    gru_gate<<<(1024 * 100 + 255) / 256, 256, 0, stream>>>(gi, gh, hp, hn, h16, tfeat, t);
    float* tmp = hp; hp = hn; hn = tmp;
  }

  // --- DGCN 1: distances via WMMA x@x^T, top-20, edge MLP via P/Q trick ---
  rowsq<<<(1024 + 255) / 256, 256, 0, stream>>>(tfeat, sq, 1024, 3200);
  gemm_nt<<<dim3(8, 8), 256, 0, stream>>>(tfeat, 3200, tfeat, 3200,
                                          xxT, 1024, 1024, 1024, 3200, nullptr, 0);
  topk_from_dot<<<1024, 32, 0, stream>>>(xxT, sq, idx1, 1024);
  transpose_f16<<<(6400 * 16 + 255) / 256, 256, 0, stream>>>(mlp1_w1, w1t, 6400, 16);
  gemm_nt<<<dim3(8, 1), 256, 0, stream>>>(tfeat, 3200, w1t, 6400,
                                          P1, 16, 1024, 16, 3200, nullptr, 0);
  gemm_nt<<<dim3(8, 1), 256, 0, stream>>>(tfeat, 3200, w1t + 3200, 6400,
                                          Q1, 16, 1024, 16, 3200, nullptr, 0);
  edge_mlp_max<<<4, 256, 0, stream>>>(P1, Q1, idx1, mlp1_b1, mlp1_w2, mlp1_b2, xdg1, 1024, 1);

  // --- pool1 + gap1 ---
  pool_score<<<4, 256, 0, stream>>>(xdg1, pool1_w, sc1, 1024);
  pool_compact<<<4, 256, 0, stream>>>(xdg1, sc1, xp1, 1024, 820);
  gap_k<<<1, 256, 0, stream>>>(xp1, 820, gate1_w, gate1_b, g1v);

  // --- DGCN 2 (tiny, scalar) ---
  topk_direct<<<820, 32, 0, stream>>>(xp1, idx2, 820);
  proj16<<<(820 + 255) / 256, 256, 0, stream>>>(xp1, mlp2_w1, P2, Q2, 820);
  edge_mlp_max<<<(820 + 255) / 256, 256, 0, stream>>>(P2, Q2, idx2, mlp2_b1, mlp2_w2, mlp2_b2,
                                                      xdg2, 820, 0);

  // --- pool2 + gap2 ---
  pool_score<<<(820 + 255) / 256, 256, 0, stream>>>(xdg2, pool2_w, sc2, 820);
  pool_compact<<<(820 + 255) / 256, 256, 0, stream>>>(xdg2, sc2, xp2, 820, 656);
  gap_k<<<1, 256, 0, stream>>>(xp2, 656, gate2_w, gate2_b, g2v);

  // --- final linear ---
  final_k<<<1, 32, 0, stream>>>(g1v, g2v, lin2_w, lin2_b, (float*)d_out);
}
